// SlidingCausalLocalSelfAttention_197568495698
// MI455X (gfx1250) — matrix operand order repaired and checked by reading, not compiled, not run on hardware
//
#include <hip/hip_runtime.h>

// ---------------------------------------------------------------------------
// Sliding causal local self-attention (b=4, n=4096, dim=1024, 16 heads, d=64,
// window=128, look_backward=1) for gfx1250: bf16 WMMA everywhere.
// ---------------------------------------------------------------------------

typedef __bf16 bf16_t;
typedef __attribute__((ext_vector_type(16))) __bf16 v16bf;
typedef __attribute__((ext_vector_type(8)))  __bf16 v8bf;
typedef __attribute__((ext_vector_type(8)))  float  v8f;

union BF16x16 { v16bf v; v8bf h[2]; };

#define WMMA_BF16(A, B, C) \
  __builtin_amdgcn_wmma_f32_16x16x32_bf16(false, (A), false, (B), (short)0, (C), false, false)

static __device__ __forceinline__ unsigned pack_bf16x2(float a, float b) {
  union { __bf16 h[2]; unsigned u; } p;
  p.h[0] = (bf16_t)a; p.h[1] = (bf16_t)b;
  return p.u;
}
static __device__ __forceinline__ unsigned pack_bf16x2h(bf16_t a, bf16_t b) {
  union { __bf16 h[2]; unsigned u; } p;
  p.h[0] = a; p.h[1] = b;
  return p.u;
}

// Sizes
#define NB      4
#define SEQ     4096
#define DIM     1024
#define HEADS   16
#define HD      64
#define WS      128
#define NWIN    32
#define BH      (NB * HEADS)
#define MROWS   (NB * SEQ)          // 16384
#define HSTRIDE ((size_t)SEQ * HD)  // per-(b,h) stride in head-major QKV

// ===========================================================================
// Kernel 1: fused QKV projection.  C = x @ W + b, written head-major bf16.
// Block: 128 threads (4 waves). Tile: 128(M) x 64(N), BK=32.
// ===========================================================================
__global__ __launch_bounds__(128) void qkv_gemm(
    const float* __restrict__ x,
    const float* __restrict__ Wq, const float* __restrict__ bq,
    const float* __restrict__ Wk, const float* __restrict__ bk,
    const float* __restrict__ Wv, const float* __restrict__ bv,
    bf16_t* __restrict__ Qh, bf16_t* __restrict__ Kh, bf16_t* __restrict__ Vh)
{
  const int which = blockIdx.z;
  const float* W    = (which == 0) ? Wq : (which == 1) ? Wk : Wv;
  const float* bias = (which == 0) ? bq : (which == 1) ? bk : bv;
  bf16_t* dst       = (which == 0) ? Qh : (which == 1) ? Kh : Vh;

  const int bm = blockIdx.x;   // 0..127 : 128-row tile
  const int bn = blockIdx.y;   // 0..15  : 64-col tile

  __shared__ bf16_t As[128][40];   // A tile, row-major, 80B rows (16B aligned)
  __shared__ bf16_t Bts[64][40];   // B tile transposed: Bts[n][k]

  const int tid  = threadIdx.x;
  const int lane = tid & 31;
  const int warp = tid >> 5;
  const int half = lane >> 4;
  const int mr   = lane & 15;

  v8f acc[2][4] = {};

  const int rowg = bm * 128 + tid;     // A-load row
  const int kp   = (tid & 15) * 2;     // B-load even k of the (k, k+1) pair
  const int ng   = (tid >> 4) * 8;     // B-load col group (8 cols)

  for (int k0 = 0; k0 < DIM; k0 += 32) {
    // --- load A tile (fp32 -> packed bf16, 64-bit LDS stores) ---
    {
      const float4* src = (const float4*)(x + (size_t)rowg * DIM + k0);
      #pragma unroll
      for (int j = 0; j < 8; ++j) {
        float4 f = src[j];
        uint2 u;
        u.x = pack_bf16x2(f.x, f.y);
        u.y = pack_bf16x2(f.z, f.w);
        *(uint2*)&As[tid][j * 4] = u;
      }
    }
    // --- load B (k,k+1) row pair, store transposed packed dwords ---
    {
      const float4* r0 = (const float4*)(W + (size_t)(k0 + kp)     * DIM + bn * 64 + ng);
      const float4* r1 = (const float4*)(W + (size_t)(k0 + kp + 1) * DIM + bn * 64 + ng);
      float4 a0 = r0[0], a1 = r0[1];
      float4 b0 = r1[0], b1 = r1[1];
      *(unsigned*)&Bts[ng + 0][kp] = pack_bf16x2(a0.x, b0.x);
      *(unsigned*)&Bts[ng + 1][kp] = pack_bf16x2(a0.y, b0.y);
      *(unsigned*)&Bts[ng + 2][kp] = pack_bf16x2(a0.z, b0.z);
      *(unsigned*)&Bts[ng + 3][kp] = pack_bf16x2(a0.w, b0.w);
      *(unsigned*)&Bts[ng + 4][kp] = pack_bf16x2(a1.x, b1.x);
      *(unsigned*)&Bts[ng + 5][kp] = pack_bf16x2(a1.y, b1.y);
      *(unsigned*)&Bts[ng + 6][kp] = pack_bf16x2(a1.z, b1.z);
      *(unsigned*)&Bts[ng + 7][kp] = pack_bf16x2(a1.w, b1.w);
    }
    if (k0 + 32 < DIM) {
      __builtin_prefetch(x + (size_t)rowg * DIM + k0 + 32, 0, 0);
      __builtin_prefetch(W + (size_t)(k0 + 32 + kp) * DIM + bn * 64 + ng, 0, 0);
    }
    __syncthreads();

    // hoist ALL fragments into distinct registers, then 8 back-to-back WMMAs
    BF16x16 afr[2], bfr[4];
    #pragma unroll
    for (int i = 0; i < 2; ++i) {
      const bf16_t* p = &As[warp * 32 + i * 16 + mr][half * 8];
      afr[i].h[0] = *(const v8bf*)p;
      afr[i].h[1] = *(const v8bf*)(p + 16);
    }
    #pragma unroll
    for (int t = 0; t < 4; ++t) {
      const bf16_t* p = &Bts[t * 16 + mr][half * 8];
      bfr[t].h[0] = *(const v8bf*)p;
      bfr[t].h[1] = *(const v8bf*)(p + 16);
    }
    #pragma unroll
    for (int t = 0; t < 4; ++t)
      #pragma unroll
      for (int i = 0; i < 2; ++i)
        acc[i][t] = WMMA_BF16(afr[i].v, bfr[t].v, acc[i][t]);
    __syncthreads();
  }

  // epilogue: + bias, scatter into head-major bf16
  #pragma unroll
  for (int i = 0; i < 2; ++i) {
    #pragma unroll
    for (int t = 0; t < 4; ++t) {
      const int col = bn * 64 + t * 16 + mr;
      const float bcol = bias[col];
      const int h  = col >> 6;
      const int dc = col & 63;
      #pragma unroll
      for (int r = 0; r < 8; ++r) {
        const int row = bm * 128 + warp * 32 + i * 16 + r + half * 8;
        const int b   = row >> 12;
        const int n   = row & 4095;
        dst[(size_t)(b * HEADS + h) * HSTRIDE + (size_t)n * HD + dc] =
            (bf16_t)(acc[i][t][r] + bcol);
      }
    }
  }
}

// ===========================================================================
// Kernel 2: sliding-window causal attention.
// One block (64 threads = 2 waves) per (bh, window, 32-row query strip).
// S held fully in registers; softmax via shfl_xor over 16-lane half groups.
// ===========================================================================
__global__ __launch_bounds__(64) void attn_kernel(
    const bf16_t* __restrict__ Qh, const bf16_t* __restrict__ Kh,
    const bf16_t* __restrict__ Vh, bf16_t* __restrict__ AO)
{
  const int blk   = blockIdx.x;       // 0..8191
  const int qquad = blk & 3;          // 32-row strip within window
  const int w     = (blk >> 2) & 31;  // window
  const int bh    = blk >> 7;         // 0..63

  const int tid  = threadIdx.x;
  const int lane = tid & 31;
  const int warp = tid >> 5;          // 0..1
  const int half = lane >> 4;
  const int mr   = lane & 15;

  __shared__ bf16_t Vt[64][264];      // V transposed: Vt[d][j], 528B rows
  __shared__ bf16_t Ps[32][264];      // normalized probabilities, bf16

  const size_t baseH = (size_t)bh * HSTRIDE;
  const int key0 = w * 128 - 128;     // global position of key j=0

  // ---- fill Vt (transpose 256x64 -> 64x256), two rows at a time so each
  //      store is a packed {V[j][d], V[j+1][d]} dword. Pad rows -> 0.
  #pragma unroll
  for (int pp = 0; pp < 2; ++pp) {
    const int j = tid * 4 + pp * 2;   // even
    const int n = key0 + j;           // pair never straddles 0 (both even)
    if (n >= 0) {
      const v8bf* s0 = (const v8bf*)(Vh + baseH + (size_t)n * HD);
      const v8bf* s1 = (const v8bf*)(Vh + baseH + (size_t)(n + 1) * HD);
      #pragma unroll
      for (int c = 0; c < 8; ++c) {
        v8bf x0 = s0[c], x1 = s1[c];
        #pragma unroll
        for (int e = 0; e < 8; ++e)
          *(unsigned*)&Vt[c * 8 + e][j] = pack_bf16x2h(x0[e], x1[e]);
      }
    } else {
      #pragma unroll
      for (int d = 0; d < 64; ++d) *(unsigned*)&Vt[d][j] = 0u;
    }
  }
  __syncthreads();

  // ---- S = Q K^T : 16 column tiles x 2 k-chunks, frags straight from global
  const int qbase = w * 128 + qquad * 32 + warp * 16;
  v8f acc[16] = {};

  BF16x16 afr[2];
  #pragma unroll
  for (int kc = 0; kc < 2; ++kc) {
    const bf16_t* qp = Qh + baseH + (size_t)(qbase + mr) * HD + kc * 32 + half * 8;
    afr[kc].h[0] = *(const v8bf*)qp;
    afr[kc].h[1] = *(const v8bf*)(qp + 16);
  }
  #pragma unroll
  for (int t = 0; t < 16; ++t) {
    if (w > 0 || t >= 8) {            // wave-uniform: skip fully padded tiles
      const int n = key0 + t * 16 + mr;
      BF16x16 bfr[2];
      #pragma unroll
      for (int kc = 0; kc < 2; ++kc) {
        const bf16_t* kpt = Kh + baseH + (size_t)n * HD + kc * 32 + half * 8;
        bfr[kc].h[0] = *(const v8bf*)kpt;
        bfr[kc].h[1] = *(const v8bf*)(kpt + 16);
      }
      #pragma unroll
      for (int kc = 0; kc < 2; ++kc)
        acc[t] = WMMA_BF16(afr[kc].v, bfr[kc].v, acc[t]);
    }
  }

  // ---- mask + scale + softmax (rows live in 16-lane half groups) ----
  const int iwin_base = qquad * 32 + warp * 16 + half * 8;  // row-in-window
  float rmax[8], rsum[8], rinv[8];
  #pragma unroll
  for (int r = 0; r < 8; ++r) rmax[r] = -3.0e38f;

  #pragma unroll
  for (int t = 0; t < 16; ++t) {
    const int j0 = t * 16 + mr;
    #pragma unroll
    for (int r = 0; r < 8; ++r) {
      const int iwin = iwin_base + r;
      const bool masked = (j0 > iwin + 128) || (w == 0 && j0 < 128);
      const float s = masked ? -3.0e38f : acc[t][r] * 0.125f;
      acc[t][r] = s;
      rmax[r] = fmaxf(rmax[r], s);
    }
  }
  #pragma unroll
  for (int m = 1; m <= 8; m <<= 1)
    #pragma unroll
    for (int r = 0; r < 8; ++r)
      rmax[r] = fmaxf(rmax[r], __shfl_xor(rmax[r], m, 32));

  #pragma unroll
  for (int r = 0; r < 8; ++r) rsum[r] = 0.0f;
  #pragma unroll
  for (int t = 0; t < 16; ++t)
    #pragma unroll
    for (int r = 0; r < 8; ++r) {
      const float p = __expf(acc[t][r] - rmax[r]);
      acc[t][r] = p;
      rsum[r] += p;
    }
  #pragma unroll
  for (int m = 1; m <= 8; m <<= 1)
    #pragma unroll
    for (int r = 0; r < 8; ++r)
      rsum[r] += __shfl_xor(rsum[r], m, 32);
  #pragma unroll
  for (int r = 0; r < 8; ++r) rinv[r] = 1.0f / rsum[r];

  const int prow = warp * 16 + half * 8;
  #pragma unroll
  for (int t = 0; t < 16; ++t)
    #pragma unroll
    for (int r = 0; r < 8; ++r)
      Ps[prow + r][t * 16 + mr] = (bf16_t)(acc[t][r] * rinv[r]);
  __syncthreads();

  // ---- O = P V : 4 output tiles x 8 k-chunks ----
  v8f acc2[4] = {};
  #pragma unroll
  for (int kc = 0; kc < 8; ++kc) {
    BF16x16 af;
    const bf16_t* ppt = &Ps[warp * 16 + mr][kc * 32 + half * 8];
    af.h[0] = *(const v8bf*)ppt;
    af.h[1] = *(const v8bf*)(ppt + 16);
    BF16x16 bfr[4];
    #pragma unroll
    for (int t2 = 0; t2 < 4; ++t2) {
      const bf16_t* vp = &Vt[t2 * 16 + mr][kc * 32 + half * 8];
      bfr[t2].h[0] = *(const v8bf*)vp;
      bfr[t2].h[1] = *(const v8bf*)(vp + 16);
    }
    #pragma unroll
    for (int t2 = 0; t2 < 4; ++t2)
      acc2[t2] = WMMA_BF16(af.v, bfr[t2].v, acc2[t2]);
  }

  // ---- scatter to (b, n, h*64+d) bf16 for the output projection ----
  const int b = bh >> 4, h = bh & 15;
  #pragma unroll
  for (int t2 = 0; t2 < 4; ++t2) {
    const int col = h * 64 + t2 * 16 + mr;
    #pragma unroll
    for (int r = 0; r < 8; ++r) {
      const int nrow = w * 128 + qquad * 32 + warp * 16 + r + half * 8;
      AO[(size_t)(b * SEQ + nrow) * DIM + col] = (bf16_t)acc2[t2][r];
    }
  }
}

// ===========================================================================
// Kernel 3: output projection.  out = AO @ Wo + bo  (fp32 out).
// ===========================================================================
__global__ __launch_bounds__(128) void out_gemm(
    const bf16_t* __restrict__ AO, const float* __restrict__ Wo,
    const float* __restrict__ bo, float* __restrict__ out)
{
  const int bm = blockIdx.x;
  const int bn = blockIdx.y;

  __shared__ bf16_t As[128][40];
  __shared__ bf16_t Bts[64][40];

  const int tid  = threadIdx.x;
  const int lane = tid & 31;
  const int warp = tid >> 5;
  const int half = lane >> 4;
  const int mr   = lane & 15;

  v8f acc[2][4] = {};

  const int rowg = bm * 128 + tid;
  const int kp   = (tid & 15) * 2;
  const int ng   = (tid >> 4) * 8;

  for (int k0 = 0; k0 < DIM; k0 += 32) {
    {  // A tile: already bf16, 4x 16B copies
      const uint4* src = (const uint4*)(AO + (size_t)rowg * DIM + k0);
      #pragma unroll
      for (int j = 0; j < 4; ++j)
        *(uint4*)&As[tid][j * 8] = src[j];
    }
    {  // B (k,k+1) row pair, store transposed packed dwords
      const float4* r0 = (const float4*)(Wo + (size_t)(k0 + kp)     * DIM + bn * 64 + ng);
      const float4* r1 = (const float4*)(Wo + (size_t)(k0 + kp + 1) * DIM + bn * 64 + ng);
      float4 a0 = r0[0], a1 = r0[1];
      float4 b0 = r1[0], b1 = r1[1];
      *(unsigned*)&Bts[ng + 0][kp] = pack_bf16x2(a0.x, b0.x);
      *(unsigned*)&Bts[ng + 1][kp] = pack_bf16x2(a0.y, b0.y);
      *(unsigned*)&Bts[ng + 2][kp] = pack_bf16x2(a0.z, b0.z);
      *(unsigned*)&Bts[ng + 3][kp] = pack_bf16x2(a0.w, b0.w);
      *(unsigned*)&Bts[ng + 4][kp] = pack_bf16x2(a1.x, b1.x);
      *(unsigned*)&Bts[ng + 5][kp] = pack_bf16x2(a1.y, b1.y);
      *(unsigned*)&Bts[ng + 6][kp] = pack_bf16x2(a1.z, b1.z);
      *(unsigned*)&Bts[ng + 7][kp] = pack_bf16x2(a1.w, b1.w);
    }
    if (k0 + 32 < DIM) {
      __builtin_prefetch(AO + (size_t)rowg * DIM + k0 + 32, 0, 0);
      __builtin_prefetch(Wo + (size_t)(k0 + 32 + kp) * DIM + bn * 64 + ng, 0, 0);
    }
    __syncthreads();

    BF16x16 afr[2], bfr[4];
    #pragma unroll
    for (int i = 0; i < 2; ++i) {
      const bf16_t* p = &As[warp * 32 + i * 16 + mr][half * 8];
      afr[i].h[0] = *(const v8bf*)p;
      afr[i].h[1] = *(const v8bf*)(p + 16);
    }
    #pragma unroll
    for (int t = 0; t < 4; ++t) {
      const bf16_t* p = &Bts[t * 16 + mr][half * 8];
      bfr[t].h[0] = *(const v8bf*)p;
      bfr[t].h[1] = *(const v8bf*)(p + 16);
    }
    #pragma unroll
    for (int t = 0; t < 4; ++t)
      #pragma unroll
      for (int i = 0; i < 2; ++i)
        acc[i][t] = WMMA_BF16(afr[i].v, bfr[t].v, acc[i][t]);
    __syncthreads();
  }

  #pragma unroll
  for (int i = 0; i < 2; ++i) {
    #pragma unroll
    for (int t = 0; t < 4; ++t) {
      const int col = bn * 64 + t * 16 + mr;
      const float bcol = bo[col];
      #pragma unroll
      for (int r = 0; r < 8; ++r) {
        const int row = bm * 128 + warp * 32 + i * 16 + r + half * 8;
        out[(size_t)row * DIM + col] = acc[i][t][r] + bcol;
      }
    }
  }
}

// ===========================================================================
extern "C" void kernel_launch(void* const* d_in, const int* in_sizes, int n_in,
                              void* d_out, int out_size, void* d_ws, size_t ws_size,
                              hipStream_t stream) {
  const float* x  = (const float*)d_in[0];
  const float* Wq = (const float*)d_in[1];
  const float* bq = (const float*)d_in[2];
  const float* Wk = (const float*)d_in[3];
  const float* bk = (const float*)d_in[4];
  const float* Wv = (const float*)d_in[5];
  const float* bv = (const float*)d_in[6];
  const float* Wo = (const float*)d_in[7];
  const float* bo = (const float*)d_in[8];

  const size_t qkvElems = (size_t)BH * SEQ * HD;   // 16.78M bf16 each
  bf16_t* Qh = (bf16_t*)d_ws;
  bf16_t* Kh = Qh + qkvElems;
  bf16_t* Vh = Kh + qkvElems;
  bf16_t* AO = Vh + qkvElems;                      // total 128 MB

  dim3 gQKV(MROWS / 128, DIM / 64, 3);
  qkv_gemm<<<gQKV, 128, 0, stream>>>(x, Wq, bq, Wk, bk, Wv, bv, Qh, Kh, Vh);

  attn_kernel<<<BH * NWIN * 4, 64, 0, stream>>>(Qh, Kh, Vh, AO);

  dim3 gOut(MROWS / 128, DIM / 64);
  out_gemm<<<gOut, 128, 0, stream>>>(AO, Wo, bo, (float*)d_out);
}
